// single_head_split_window_attention_44315472560365
// MI455X (gfx1250) — matrix operand order repaired and checked by reading, not compiled, not run on hardware
//
#include <hip/hip_runtime.h>

typedef _Float16 v16h __attribute__((ext_vector_type(16)));
typedef _Float16 v8h  __attribute__((ext_vector_type(8)));
typedef float    v8f  __attribute__((ext_vector_type(8)));
typedef float    f4   __attribute__((ext_vector_type(4)));

#define NSQ   64      // ns*ns windows
#define Wdim  256
#define Cdim  128
#define LW    1024    // tokens per window (32*32)
#define SHIFT 16
#define INV_SCALE 0.08838834764831845f  // 1/sqrt(128)

union V16 { v16h v; v8h h[2]; _Float16 e[16]; };

__global__ __launch_bounds__(128) void swin_attn_wmma_kernel(
    const float* __restrict__ q, const float* __restrict__ k,
    const float* __restrict__ v, const float* __restrict__ mask,
    float* __restrict__ out)
{
  __shared__ _Float16 Vlds[Cdim * 32];      // V chunk transposed: [ch][key]
  __shared__ _Float16 Plds[4][16 * 32];     // per-wave probability tile

  const int tid     = threadIdx.x;
  const int wave    = tid >> 5;
  const int lane    = tid & 31;
  const int lane_lo = lane & 15;
  const int half    = lane >> 4;

  const int w_id = blockIdx.x >> 4;   // window 0..63
  const int qt   = blockIdx.x & 15;   // 64-row q tile within window
  const int wy = w_id >> 3, wx = w_id & 7;
  const int q_base = qt * 64 + wave * 16;

  // rolled token address: h unshifted, w shifted by +SHIFT (source roll -SHIFT)
  auto tok_of = [&](int local) -> int {
    int ly = local >> 5, lx = local & 31;
    int hh = wy * 32 + ly;
    int ww = (wx * 32 + lx + SHIFT) & (Wdim - 1);
    return hh * Wdim + ww;
  };

  // ---------- Q fragments (A layout), kept in registers for whole loop ----------
  V16 qfrag[4];
  {
    const float* qp = q + (size_t)tok_of(q_base + lane_lo) * Cdim;
    for (int f = 0; f < 4; ++f) {
      for (int g = 0; g < 2; ++g) {
        int chb = 32 * f + 16 * half + 8 * g;
        int src = (chb + SHIFT) & (Cdim - 1);       // channel roll
        f4 a = *(const f4*)(qp + src);
        f4 b = *(const f4*)(qp + src + 4);
        int e0 = 8 * g;
        for (int e = 0; e < 4; ++e) {
          qfrag[f].e[e0 + e]     = (_Float16)a[e];
          qfrag[f].e[e0 + 4 + e] = (_Float16)b[e];
        }
      }
    }
  }

  float m_run[8], l_run[8], pscale[8];
  v8f acc[8];
  for (int r = 0; r < 8; ++r) { m_run[r] = -1e30f; l_run[r] = 0.f; }
  for (int t = 0; t < 8; ++t)
    for (int i = 0; i < 8; ++i) acc[t][i] = 0.f;

  const float* mwin = mask + (size_t)w_id * LW * LW;

  for (int j0 = 0; j0 < LW; j0 += 32) {
    __syncthreads();
    // ---------- stage 32-key V chunk into LDS, transposed, f16 ----------
    {
      int key = tid >> 2;              // 0..31
      int chb = (tid & 3) * 32;        // 0,32,64,96
      const float* vp = v + (size_t)tok_of(j0 + key) * Cdim;
      for (int g = 0; g < 8; ++g) {
        int ch  = chb + 4 * g;
        int src = (ch + SHIFT) & (Cdim - 1);
        f4 vv = *(const f4*)(vp + src);
        for (int e = 0; e < 4; ++e)
          Vlds[(ch + e) * 32 + key] = (_Float16)vv[e];
      }
    }
    __syncthreads();

    // ---------- scores: two 16-key tiles, K=128 via 4 chained WMMAs ----------
    v8f s[2];
    for (int t = 0; t < 2; ++t) {
      v8f cacc; for (int i = 0; i < 8; ++i) cacc[i] = 0.f;
      const float* kp = k + (size_t)tok_of(j0 + 16 * t + lane_lo) * Cdim;
      for (int f = 0; f < 4; ++f) {
        V16 kf;                         // B layout: 16 contiguous channels / lane half
        int src = (32 * f + 16 * half + SHIFT) & (Cdim - 1);
        for (int g = 0; g < 4; ++g) {
          f4 a = *(const f4*)(kp + src + 4 * g);
          for (int e = 0; e < 4; ++e) kf.e[4 * g + e] = (_Float16)a[e];
        }
        cacc = __builtin_amdgcn_wmma_f32_16x16x32_f16(
            false, qfrag[f].v, false, kf.v, (short)0, cacc, false, false);
      }
      for (int r = 0; r < 8; ++r) {
        int qrow = q_base + r + 8 * half;
        float mm = mwin[(size_t)qrow * LW + (j0 + 16 * t + lane_lo)];
        cacc[r] = cacc[r] * INV_SCALE + mm;
      }
      s[t] = cacc;
    }

    // ---------- online softmax (rows span 16 lanes within a half-wave) ----------
    for (int r = 0; r < 8; ++r) {
      float tmax = fmaxf(s[0][r], s[1][r]);
      for (int off = 1; off < 16; off <<= 1)
        tmax = fmaxf(tmax, __shfl_xor(tmax, off, 32));
      float mnew = fmaxf(m_run[r], tmax);
      float sc   = __expf(m_run[r] - mnew);
      float p0   = __expf(s[0][r] - mnew);
      float p1   = __expf(s[1][r] - mnew);
      s[0][r] = p0; s[1][r] = p1;
      float rs = p0 + p1;
      for (int off = 1; off < 16; off <<= 1)
        rs += __shfl_xor(rs, off, 32);
      l_run[r]  = l_run[r] * sc + rs;
      m_run[r]  = mnew;
      pscale[r] = sc;
    }
    for (int t = 0; t < 8; ++t)
      for (int r = 0; r < 8; ++r)
        acc[t][r] *= pscale[r];

    // ---------- P: C-layout -> A-layout via wave-private LDS ----------
    _Float16* P = Plds[wave];
    for (int t = 0; t < 2; ++t)
      for (int r = 0; r < 8; ++r)
        P[(r + 8 * half) * 32 + 16 * t + lane_lo] = (_Float16)s[t][r];
    asm volatile("s_wait_dscnt 0" ::: "memory");   // same-wave DS ops in-order

    V16 pf;
    pf.h[0] = *(const v8h*)&P[lane_lo * 32 + 8 * half];
    pf.h[1] = *(const v8h*)&P[lane_lo * 32 + 16 + 8 * half];

    // ---------- out += P @ V over 8 channel tiles ----------
    for (int t = 0; t < 8; ++t) {
      V16 vf;
      int ch = 16 * t + lane_lo;
      vf.h[0] = *(const v8h*)&Vlds[ch * 32 + 8 * half];
      vf.h[1] = *(const v8h*)&Vlds[ch * 32 + 16 + 8 * half];
      acc[t] = __builtin_amdgcn_wmma_f32_16x16x32_f16(
          false, pf.v, false, vf.v, (short)0, acc[t], false, false);
    }
  }

  // ---------- epilogue: normalize and scatter with rolled addressing ----------
  for (int r = 0; r < 8; ++r) {
    int tok = tok_of(q_base + r + 8 * half);
    float inv_l = 1.f / l_run[r];
    for (int t = 0; t < 8; ++t) {
      int ch = (16 * t + lane_lo + SHIFT) & (Cdim - 1);
      out[(size_t)tok * Cdim + ch] = acc[t][r] * inv_l;
    }
  }
}

extern "C" void kernel_launch(void* const* d_in, const int* in_sizes, int n_in,
                              void* d_out, int out_size, void* d_ws, size_t ws_size,
                              hipStream_t stream) {
  (void)in_sizes; (void)n_in; (void)out_size; (void)d_ws; (void)ws_size;
  const float* q    = (const float*)d_in[0];
  const float* k    = (const float*)d_in[1];
  const float* v    = (const float*)d_in[2];
  const float* mask = (const float*)d_in[3];
  float* out = (float*)d_out;
  dim3 grid(64 * 16), block(128);
  hipLaunchKernelGGL(swin_attn_wmma_kernel, grid, block, 0, stream,
                     q, k, v, mask, out);
}
